// Speaker_64390149701892
// MI455X (gfx1250) — matrix operand order repaired and checked by reading, not compiled, not run on hardware
//
#include <hip/hip_runtime.h>
#include <stdint.h>

#define B_SZ    16384
#define E_SZ    256
#define V_SZ    512
#define T_MAX   16
#define ROWS    64                  // rows per block (4 waves x 16 rows)
#define XPITCH  264                 // padded LDS row pitch (halfs): bank-conflict-free A loads
#define HSZ     (ROWS * XPITCH)
#define WIH_OFF  0
#define WHH_OFF  196608             // 768*256
#define WOUT_OFF 393216             // + 768*256
#define WD2E_OFF 524288             // + 512*256
#define WS_ELEMS 655360             // + 512*256

typedef __attribute__((ext_vector_type(16))) __bf16 v16bf;
typedef __attribute__((ext_vector_type(8)))  float  v8f;

union FragBF { uint4 u[2]; v16bf v; };

__device__ __forceinline__ uint16_t f2bf(float f) {
  uint32_t u = __float_as_uint(f);
  u += 0x7FFFu + ((u >> 16) & 1u);          // round-to-nearest-even
  return (uint16_t)(u >> 16);
}
__device__ __forceinline__ float bf2f(uint16_t h) {
  return __uint_as_float(((uint32_t)h) << 16);
}
__device__ __forceinline__ float sigmoidf_(float x) { return 1.0f / (1.0f + __expf(-x)); }
__device__ __forceinline__ float u01hash(uint32_t a, uint32_t b) {
  uint32_t x = a * 0x9E3779B9u ^ (b + 0x7F4A7C15u) * 0x85EBCA6Bu;
  x ^= x >> 16; x *= 0x7FEB352Du; x ^= x >> 15; x *= 0x846CA68Bu; x ^= x >> 16;
  return (float)(x >> 8) * (1.0f / 16777216.0f) + 5.9604645e-8f;
}

// ---------- prep: fp32 weights/tables -> bf16 in workspace ----------
__global__ void prep_bf16(const float* __restrict__ wih, const float* __restrict__ whh,
                          const float* __restrict__ wout, const float* __restrict__ d2e,
                          uint16_t* __restrict__ ws) {
  int i = blockIdx.x * blockDim.x + threadIdx.x;
  if (i < WHH_OFF)            ws[i] = f2bf(wih[i]);
  else if (i < WOUT_OFF)      ws[i] = f2bf(whh[i - WHH_OFF]);
  else if (i < WD2E_OFF)      ws[i] = f2bf(wout[i - WOUT_OFF]);
  else if (i < WS_ELEMS)      ws[i] = f2bf(d2e[i - WD2E_OFF]);
}

#define WMMA_BF16(C, A, B) \
  (C) = __builtin_amdgcn_wmma_f32_16x16x32_bf16(false, (A), false, (B), (short)0, (C), false, false)

// ---------- main: full 16-step GRU decode, one wave owns 16 rows ----------
__global__ __launch_bounds__(128, 1)
void speaker_gru_wmma(const int* __restrict__ actions,
                      const float* __restrict__ h1_emb,
                      const float* __restrict__ b_ih,
                      const float* __restrict__ b_hh,
                      const float* __restrict__ b_out,
                      const uint16_t* __restrict__ ws,
                      float* __restrict__ out) {
  extern __shared__ uint16_t smem[];
  const int tid  = threadIdx.x;
  const int wave = tid >> 5;
  const int lane = tid & 31;
  const int half = lane >> 4;
  const int n16  = lane & 15;
  const int wrow = wave * 16;                 // first row within block owned by this wave
  const int blkrow = blockIdx.x * ROWS;

  const int HOFF_A = 0;
  const int HOFF_B = HSZ;
  const int XOFF   = 2 * HSZ;
  const int TOKOFF = 3 * HSZ;

  // init h = h1_emb[actions[row]]  (fp32 -> bf16 into buffer A)
  for (int idx = lane; idx < 16 * (E_SZ / 8); idx += 32) {
    int r  = idx >> 5;
    int kc = (idx & 31) * 8;
    int act = actions[blkrow + wrow + r];
    const float* src = h1_emb + (size_t)act * E_SZ + kc;
    int dst = HOFF_A + (wrow + r) * XPITCH + kc;
#pragma unroll
    for (int j = 0; j < 8; ++j) smem[dst + j] = f2bf(src[j]);
  }
  // init x = d2e_emb[token 0] (already bf16 in ws)
  for (int idx = lane; idx < 16 * (E_SZ / 8); idx += 32) {
    int r  = idx >> 5;
    int kc = (idx & 31) * 8;
    *(uint4*)&smem[XOFF + (wrow + r) * XPITCH + kc] = *(const uint4*)(ws + WD2E_OFF + kc);
  }
  __syncthreads();

  // persistent per-lane state for rows wrow + half*8 + e  (replicated across 16 lanes)
  float logp[8]; int Ncnt[8]; int alv[8];
#pragma unroll
  for (int e = 0; e < 8; ++e) { logp[e] = 0.0f; Ncnt[e] = T_MAX; alv[e] = 1; }

#pragma unroll 1
  for (int t = 0; t < T_MAX; ++t) {
    const int hOld = (t & 1) ? HOFF_B : HOFF_A;
    const int hNew = (t & 1) ? HOFF_A : HOFF_B;

    // ===== GRU gates: 16-wide column chunks of E =====
#pragma unroll 1
    for (int c = 0; c < E_SZ / 16; ++c) {
      v8f air = {}, aiz = {}, ain = {}, ahr = {}, ahz = {}, ahn = {};
#pragma unroll
      for (int kc = 0; kc < E_SZ / 32; ++kc) {
        const int kb = kc * 32;
        FragBF fx, fh;
        {
          int ia = XOFF + (wrow + n16) * XPITCH + kb + half * 16;
          fx.u[0] = *(const uint4*)&smem[ia];
          fx.u[1] = *(const uint4*)&smem[ia + 8];
          int ih = hOld + (wrow + n16) * XPITCH + kb + half * 16;
          fh.u[0] = *(const uint4*)&smem[ih];
          fh.u[1] = *(const uint4*)&smem[ih + 8];
        }
        const int colrow = c * 16 + n16;   // weight output row this lane supplies
        const int kofs   = kb + half * 16; // B k order used here (same as A): lanes 0-15 K=0..15, 16-31 K=16..31
        const uint16_t* pir = ws + WIH_OFF  + (size_t)(0 * E_SZ + colrow) * E_SZ + kofs;
        const uint16_t* piz = ws + WIH_OFF  + (size_t)(1 * E_SZ + colrow) * E_SZ + kofs;
        const uint16_t* pin = ws + WIH_OFF  + (size_t)(2 * E_SZ + colrow) * E_SZ + kofs;
        const uint16_t* phr = ws + WHH_OFF  + (size_t)(0 * E_SZ + colrow) * E_SZ + kofs;
        const uint16_t* phz = ws + WHH_OFF  + (size_t)(1 * E_SZ + colrow) * E_SZ + kofs;
        const uint16_t* phn = ws + WHH_OFF  + (size_t)(2 * E_SZ + colrow) * E_SZ + kofs;
        FragBF bir, biz, bin, bhr, bhz, bhn;
        bir.u[0] = *(const uint4*)pir; bir.u[1] = *(const uint4*)(pir + 8);
        biz.u[0] = *(const uint4*)piz; biz.u[1] = *(const uint4*)(piz + 8);
        bin.u[0] = *(const uint4*)pin; bin.u[1] = *(const uint4*)(pin + 8);
        bhr.u[0] = *(const uint4*)phr; bhr.u[1] = *(const uint4*)(phr + 8);
        bhz.u[0] = *(const uint4*)phz; bhz.u[1] = *(const uint4*)(phz + 8);
        bhn.u[0] = *(const uint4*)phn; bhn.u[1] = *(const uint4*)(phn + 8);
        WMMA_BF16(air, fx.v, bir.v);
        WMMA_BF16(aiz, fx.v, biz.v);
        WMMA_BF16(ain, fx.v, bin.v);
        WMMA_BF16(ahr, fh.v, bhr.v);
        WMMA_BF16(ahz, fh.v, bhz.v);
        WMMA_BF16(ahn, fh.v, bhn.v);
      }
      const int col = c * 16 + n16;
      const float vbir = b_ih[col], vbiz = b_ih[E_SZ + col], vbin = b_ih[2 * E_SZ + col];
      const float vbhr = b_hh[col], vbhz = b_hh[E_SZ + col], vbhn = b_hh[2 * E_SZ + col];
#pragma unroll
      for (int e = 0; e < 8; ++e) {          // C layout: VGPR e -> M = e + half*8
        const int row = wrow + half * 8 + e;
        float r = sigmoidf_(air[e] + vbir + ahr[e] + vbhr);
        float z = sigmoidf_(aiz[e] + vbiz + ahz[e] + vbhz);
        float n = tanhf(ain[e] + vbin + r * (ahn[e] + vbhn));
        float hp = bf2f(smem[hOld + row * XPITCH + col]);
        smem[hNew + row * XPITCH + col] = f2bf((1.0f - z) * n + z * hp);
      }
    }

    // ===== logits + gumbel sampling + online logsumexp =====
    float runM[8], runS[8], bKey[8], bVal[8]; int bTok[8];
#pragma unroll
    for (int e = 0; e < 8; ++e) {
      runM[e] = -__builtin_inff(); runS[e] = 0.0f;
      bKey[e] = -__builtin_inff(); bVal[e] = 0.0f; bTok[e] = 0;
    }

#pragma unroll 1
    for (int vt = 0; vt < V_SZ / 16; ++vt) {
      v8f acc = {};
#pragma unroll
      for (int kc = 0; kc < E_SZ / 32; ++kc) {
        const int kb = kc * 32;
        FragBF fh, bw;
        int ih = hNew + (wrow + n16) * XPITCH + kb + half * 16;
        fh.u[0] = *(const uint4*)&smem[ih];
        fh.u[1] = *(const uint4*)&smem[ih + 8];
        const uint16_t* pw = ws + WOUT_OFF + (size_t)(vt * 16 + n16) * E_SZ + kb + half * 16;
        bw.u[0] = *(const uint4*)pw; bw.u[1] = *(const uint4*)(pw + 8);
        WMMA_BF16(acc, fh.v, bw.v);
      }
      const int col = vt * 16 + n16;
      const float bo = b_out[col];
      float val[8], key[8], ex[8], tmax[8]; int vtok[8];
#pragma unroll
      for (int e = 0; e < 8; ++e) {
        const int grow = blkrow + wrow + half * 8 + e;
        float v = acc[e] + bo;
        float u = u01hash((uint32_t)(grow * T_MAX + t), (uint32_t)col);
        val[e] = v; vtok[e] = col; tmax[e] = v;
        key[e] = v - __logf(-__logf(u));     // gumbel-max key
      }
#pragma unroll
      for (int mk = 1; mk < 16; mk <<= 1)
#pragma unroll
        for (int e = 0; e < 8; ++e)
          tmax[e] = fmaxf(tmax[e], __shfl_xor(tmax[e], mk, 32));
#pragma unroll
      for (int e = 0; e < 8; ++e) ex[e] = __expf(val[e] - tmax[e]);
#pragma unroll
      for (int mk = 1; mk < 16; mk <<= 1) {
#pragma unroll
        for (int e = 0; e < 8; ++e) {
          ex[e] += __shfl_xor(ex[e], mk, 32);
          float ok = __shfl_xor(key[e], mk, 32);
          float ov = __shfl_xor(val[e], mk, 32);
          int   ot = __shfl_xor(vtok[e], mk, 32);
          if (ok > key[e]) { key[e] = ok; val[e] = ov; vtok[e] = ot; }
        }
      }
#pragma unroll
      for (int e = 0; e < 8; ++e) {
        float nm = fmaxf(runM[e], tmax[e]);
        runS[e] = runS[e] * __expf(runM[e] - nm) + ex[e] * __expf(tmax[e] - nm);
        runM[e] = nm;
        if (key[e] > bKey[e]) { bKey[e] = key[e]; bVal[e] = val[e]; bTok[e] = vtok[e]; }
      }
    }

    // finalize step t per row (all 16 lanes hold identical reduced values)
#pragma unroll
    for (int e = 0; e < 8; ++e) {
      const int row  = wrow + half * 8 + e;
      const int grow = blkrow + row;
      float lp = bVal[e] - runM[e] - __logf(runS[e]);
      if (alv[e]) logp[e] += lp;
      int tok = alv[e] ? bTok[e] : 0;
      if (alv[e] && tok == 0) Ncnt[e] = t + 1;
      alv[e] = alv[e] && (tok != 0);
      if (n16 == 0) {
        out[(size_t)grow * T_MAX + t] = (float)tok;
        smem[TOKOFF + row] = (uint16_t)tok;
      }
    }
    __syncthreads();

    // gather next-step embeddings: x = d2e[tok] (bf16 -> LDS)
    for (int idx = lane; idx < 16 * (E_SZ / 8); idx += 32) {
      int r  = idx >> 5;
      int kc = (idx & 31) * 8;
      int tok = smem[TOKOFF + wrow + r];
      *(uint4*)&smem[XOFF + (wrow + r) * XPITCH + kc] =
          *(const uint4*)(ws + WD2E_OFF + (size_t)tok * E_SZ + kc);
    }
    __syncthreads();
  }

  // outputs: N then logp (tuple order after utterance)
  if (n16 == 0) {
#pragma unroll
    for (int e = 0; e < 8; ++e) {
      const int grow = blkrow + wrow + half * 8 + e;
      out[(size_t)B_SZ * T_MAX + grow]        = (float)Ncnt[e];
      out[(size_t)B_SZ * T_MAX + B_SZ + grow] = logp[e];
    }
  }
}

extern "C" void kernel_launch(void* const* d_in, const int* in_sizes, int n_in,
                              void* d_out, int out_size, void* d_ws, size_t ws_size,
                              hipStream_t stream) {
  const int*   actions = (const int*)  d_in[0];
  // d_in[1] = global_idxes (bookkeeping only, unused)
  const float* h1_emb  = (const float*)d_in[2];
  const float* d2e     = (const float*)d_in[3];
  const float* W_ih    = (const float*)d_in[4];
  const float* W_hh    = (const float*)d_in[5];
  const float* b_ih    = (const float*)d_in[6];
  const float* b_hh    = (const float*)d_in[7];
  const float* W_out   = (const float*)d_in[8];
  const float* b_out   = (const float*)d_in[9];
  uint16_t* ws  = (uint16_t*)d_ws;
  float*    out = (float*)d_out;

  prep_bf16<<<(WS_ELEMS + 255) / 256, 256, 0, stream>>>(W_ih, W_hh, W_out, d2e, ws);

  const size_t smem_bytes = (size_t)(3 * HSZ + ROWS) * sizeof(uint16_t); // ~101.5 KB of the 320 KB LDS
  speaker_gru_wmma<<<B_SZ / ROWS, 128, smem_bytes, stream>>>(
      actions, h1_emb, b_ih, b_hh, b_out, ws, out);
}
